// GatedAttentionTransformer_66322884985394
// MI455X (gfx1250) — matrix operand order repaired and checked by reading, not compiled, not run on hardware
//
#include <hip/hip_runtime.h>
#include <hip/hip_bf16.h>

typedef __attribute__((ext_vector_type(16))) __bf16 v16bf;
typedef __attribute__((ext_vector_type(8)))  float  v8f;

#define BM 128
#define BN 128
#define BK 32
#define LDSPAD 8          // pad row stride to 40 bf16 (80B) -> conflict-free 16-row access
#define EPI_NONE 0
#define EPI_SILU 1
#define EPI_SQRELU_CAUSAL 2

// ---------------------------------------------------------------------------
// Generic batched GEMM: C[z] = epi( A[z] (MxK) @ op(B[z]) (KxN) * scale ) [+ Add[z]]
// A,B fp32 in HBM, converted to bf16 while staging into LDS; fp32 accumulation
// via v_wmma_f32_16x16x32_bf16. Requires M%128==0, N%128==0, K%32==0.
// ---------------------------------------------------------------------------
__global__ __launch_bounds__(256) void gemm_bf16_kernel(
    const float* __restrict__ A, const float* __restrict__ B,
    float* __restrict__ C, const float* __restrict__ Add,
    int M, int N, int K, int transB, int epi, float scale,
    long long strideA, long long strideB, long long strideC)
{
    __shared__ __bf16 As[BM][BK + LDSPAD];   // (row, k)
    __shared__ __bf16 Bs[BN][BK + LDSPAD];   // (col, k)  (B stored K-contiguous per column)

    const int tid  = threadIdx.x;
    const int lane = tid & 31;
    const int wave = tid >> 5;
    const int mrow = lane & 15;     // row/col within 16x16 tile
    const int kg   = lane >> 4;     // K-group select per ISA A/B layouts
    const int waveM = (wave & 3) * 32;   // 4 waves tile M
    const int waveN = (wave >> 2) * 64;  // 2 waves tile N
    const int blockM = blockIdx.y * BM;
    const int blockN = blockIdx.x * BN;
    const int z = blockIdx.z;

    A += (long long)z * strideA;
    B += (long long)z * strideB;
    C += (long long)z * strideC;
    if (Add) Add += (long long)z * strideC;

    v8f acc[2][4];
#pragma unroll
    for (int mi = 0; mi < 2; ++mi)
#pragma unroll
        for (int nj = 0; nj < 4; ++nj) acc[mi][nj] = (v8f){0,0,0,0,0,0,0,0};

    for (int k0 = 0; k0 < K; k0 += BK) {
        // ---- stage A tile: 128x32 fp32 -> bf16 ----
        {
            const int row  = tid >> 1;
            const int half = tid & 1;
            const float* src = A + (long long)(blockM + row) * K + k0 + half * 16;
            __bf16* dst = &As[row][half * 16];
#pragma unroll
            for (int j = 0; j < 4; ++j) {
                float4 f = reinterpret_cast<const float4*>(src)[j];
                dst[4*j+0] = (__bf16)f.x; dst[4*j+1] = (__bf16)f.y;
                dst[4*j+2] = (__bf16)f.z; dst[4*j+3] = (__bf16)f.w;
            }
        }
        // ---- stage B tile into (col,k) layout ----
        if (!transB) {
            // B is KxN row-major: coalesced float4 reads along N, transposed bf16 writes
            const int n4  = tid & 31;        // 32 groups of 4 columns
            const int kk0 = (tid >> 5) * 4;  // 8 groups of 4 k-rows
#pragma unroll
            for (int kk = 0; kk < 4; ++kk) {
                float4 f = *reinterpret_cast<const float4*>(
                    B + (long long)(k0 + kk0 + kk) * N + blockN + n4 * 4);
                Bs[n4*4+0][kk0+kk] = (__bf16)f.x;
                Bs[n4*4+1][kk0+kk] = (__bf16)f.y;
                Bs[n4*4+2][kk0+kk] = (__bf16)f.z;
                Bs[n4*4+3][kk0+kk] = (__bf16)f.w;
            }
        } else {
            // B is NxK row-major (e.g. k-matrix for q@k^T): direct row copy
            const int row  = tid >> 1;
            const int half = tid & 1;
            const float* src = B + (long long)(blockN + row) * K + k0 + half * 16;
            __bf16* dst = &Bs[row][half * 16];
#pragma unroll
            for (int j = 0; j < 4; ++j) {
                float4 f = reinterpret_cast<const float4*>(src)[j];
                dst[4*j+0] = (__bf16)f.x; dst[4*j+1] = (__bf16)f.y;
                dst[4*j+2] = (__bf16)f.z; dst[4*j+3] = (__bf16)f.w;
            }
        }
        __syncthreads();

        // ---- load fragments per CDNA5 WMMA VGPR layouts ----
        // A 16x32 bf16: lane(0-15)=row M, kg selects K-octet; VGPR0-3 = K[kg*8..+7],
        // VGPR4-7 = K[kg*8+16..+23]
        v16bf af[2];
#pragma unroll
        for (int mi = 0; mi < 2; ++mi) {
            const __bf16* p = &As[waveM + mi*16 + mrow][kg * 8];
            uint4* d = reinterpret_cast<uint4*>(&af[mi]);
            d[0] = *reinterpret_cast<const uint4*>(p);
            d[1] = *reinterpret_cast<const uint4*>(p + 16);
        }
        // B 32x16 bf16: lane(0-15)=col N, kg selects K-octet; K[kg*8..+7], K[kg*8+16..+23]
        v16bf bfr[4];
#pragma unroll
        for (int nj = 0; nj < 4; ++nj) {
            const __bf16* p = &Bs[waveN + nj*16 + mrow][kg * 8];
            uint4* d = reinterpret_cast<uint4*>(&bfr[nj]);
            d[0] = *reinterpret_cast<const uint4*>(p);
            d[1] = *reinterpret_cast<const uint4*>(p + 16);
        }

#pragma unroll
        for (int mi = 0; mi < 2; ++mi)
#pragma unroll
            for (int nj = 0; nj < 4; ++nj)
                acc[mi][nj] = __builtin_amdgcn_wmma_f32_16x16x32_bf16(
                    false, af[mi], false, bfr[nj], (short)0, acc[mi][nj], false, false);
        __syncthreads();
    }

    // ---- epilogue: C/D layout -> lane(0-15)=col, VGPR p = row p (+8 for lanes 16-31)
#pragma unroll
    for (int mi = 0; mi < 2; ++mi)
#pragma unroll
        for (int nj = 0; nj < 4; ++nj)
#pragma unroll
            for (int p = 0; p < 8; ++p) {
                const int row = blockM + waveM + mi*16 + p + (kg << 3);
                const int col = blockN + waveN + nj*16 + mrow;
                float v = acc[mi][nj][p] * scale;
                if (epi == EPI_SILU) {
                    v = v / (1.0f + __expf(-v));
                } else if (epi == EPI_SQRELU_CAUSAL) {
                    v = (col <= row) ? fmaxf(v, 0.0f) : 0.0f;
                    v = v * v;
                }
                const long long off = (long long)row * N + col;
                if (Add) v += Add[off];
                C[off] = v;
            }
}

// ---------------------------------------------------------------------------
// Embedding gather: one block per row, float4 copy of D=1024
// ---------------------------------------------------------------------------
__global__ __launch_bounds__(256) void embed_kernel(
    const int* __restrict__ tokens, const float* __restrict__ emb,
    float* __restrict__ x, int D)
{
    const int row = blockIdx.x;
    const long long src = (long long)tokens[row] * D;
    const float4* s = reinterpret_cast<const float4*>(emb + src);
    float4* d = reinterpret_cast<float4*>(x + (long long)row * D);
    for (int i = threadIdx.x; i < (D >> 2); i += blockDim.x) d[i] = s[i];
}

// ---------------------------------------------------------------------------
// LayerNorm over last dim (D=1024), one block (256 thr) per row
// ---------------------------------------------------------------------------
__global__ __launch_bounds__(256) void ln_kernel(
    const float* __restrict__ x, const float* __restrict__ g,
    const float* __restrict__ b, float* __restrict__ y, int D)
{
    __shared__ float red0[256], red1[256];
    const int tid = threadIdx.x;
    const float* xr = x + (long long)blockIdx.x * D;
    float s = 0.f, s2 = 0.f;
    for (int i = tid; i < D; i += blockDim.x) { float v = xr[i]; s += v; s2 += v*v; }
    red0[tid] = s; red1[tid] = s2;
    __syncthreads();
    for (int off = 128; off > 0; off >>= 1) {
        if (tid < off) { red0[tid] += red0[tid+off]; red1[tid] += red1[tid+off]; }
        __syncthreads();
    }
    const float mu  = red0[0] / D;
    const float var = red1[0] / D - mu * mu;
    const float inv = rsqrtf(var + 1e-5f);
    float* yr = y + (long long)blockIdx.x * D;
    for (int i = tid; i < D; i += blockDim.x)
        yr[i] = (xr[i] - mu) * inv * g[i] + b[i];
}

// ---------------------------------------------------------------------------
// RoPE in place on (BT, S=128): thread j handles pair (j, j+64); t = row % T
// ---------------------------------------------------------------------------
__global__ __launch_bounds__(64) void rope_kernel(float* __restrict__ qk, int T, int S)
{
    const int row = blockIdx.x;
    const int j = threadIdx.x;              // 0..63
    const int t = row % T;
    const float inv = __expf(-((float)(2 * j) / (float)S) * 9.2103403719761836f); // ln(10000)
    const float ang = (float)t * inv;
    float c, s;
    __sincosf(ang, &s, &c);
    float* p = qk + (long long)row * S;
    const float x1 = p[j], x2 = p[j + 64];
    p[j]      = x1 * c - x2 * s;
    p[j + 64] = x1 * s + x2 * c;
}

// ---------------------------------------------------------------------------
// Gate: g = u * av elementwise (float4)
// ---------------------------------------------------------------------------
__global__ __launch_bounds__(256) void gate_kernel(
    const float* __restrict__ u, const float* __restrict__ av,
    float* __restrict__ g, long long n4)
{
    long long i = (long long)blockIdx.x * blockDim.x + threadIdx.x;
    if (i < n4) {
        float4 a = reinterpret_cast<const float4*>(u)[i];
        float4 b = reinterpret_cast<const float4*>(av)[i];
        float4 r; r.x = a.x*b.x; r.y = a.y*b.y; r.z = a.z*b.z; r.w = a.w*b.w;
        reinterpret_cast<float4*>(g)[i] = r;
    }
}

extern "C" void kernel_launch(void* const* d_in, const int* in_sizes, int n_in,
                              void* d_out, int out_size, void* d_ws, size_t ws_size,
                              hipStream_t stream)
{
    (void)in_sizes; (void)n_in; (void)out_size; (void)ws_size;
    constexpr int Vv = 32000, D = 1024, E = 2048, S = 128, L = 4;
    constexpr int Bb = 2, T = 2048, BT = Bb * T;

    const int*   tokens = (const int*)  d_in[0];
    const float* emb    = (const float*)d_in[1];
    const float* Wu     = (const float*)d_in[2];
    const float* Wv     = (const float*)d_in[3];
    const float* Wq     = (const float*)d_in[4];
    const float* Wk     = (const float*)d_in[5];
    const float* Wo     = (const float*)d_in[6];
    const float* ln_g   = (const float*)d_in[7];
    const float* ln_b   = (const float*)d_in[8];
    const float* lnf_g  = (const float*)d_in[9];
    const float* lnf_b  = (const float*)d_in[10];
    const float* Wlm    = (const float*)d_in[11];
    float* out = (float*)d_out;

    // workspace carve-up (floats)
    float* ws   = (float*)d_ws;
    float* x    = ws;                       // BT*D
    float* xn   = x    + (long long)BT * D; // BT*D
    float* u    = xn   + (long long)BT * D; // BT*E
    float* v    = u    + (long long)BT * E; // BT*E
    float* q    = v    + (long long)BT * E; // BT*S
    float* k    = q    + (long long)BT * S; // BT*S
    float* attn = k    + (long long)BT * S; // B*T*T
    float* av   = attn + (long long)Bb * T * T; // BT*E
    float* g    = av   + (long long)BT * E; // BT*E

    const float attn_scale = 0.08838834764831845f; // 1/sqrt(128)
    const dim3 blk(256);

    embed_kernel<<<BT, blk, 0, stream>>>(tokens, emb, x, D);

    for (int i = 0; i < L; ++i) {
        const float* Wui = Wu + (long long)i * D * E;
        const float* Wvi = Wv + (long long)i * D * E;
        const float* Wqi = Wq + (long long)i * E * S;
        const float* Wki = Wk + (long long)i * E * S;
        const float* Woi = Wo + (long long)i * E * D;

        ln_kernel<<<BT, blk, 0, stream>>>(x, ln_g + (long long)i * D, ln_b + (long long)i * D, xn, D);

        // u = silu(xn @ Wu), v = silu(xn @ Wv) : (BT, E, D)
        gemm_bf16_kernel<<<dim3(E/BN, BT/BM, 1), blk, 0, stream>>>(
            xn, Wui, u, nullptr, BT, E, D, 0, EPI_SILU, 1.0f, 0, 0, 0);
        gemm_bf16_kernel<<<dim3(E/BN, BT/BM, 1), blk, 0, stream>>>(
            xn, Wvi, v, nullptr, BT, E, D, 0, EPI_SILU, 1.0f, 0, 0, 0);

        // q = u @ Wq, k = v @ Wk : (BT, S, E)
        gemm_bf16_kernel<<<dim3(S/BN, BT/BM, 1), blk, 0, stream>>>(
            u, Wqi, q, nullptr, BT, S, E, 0, EPI_NONE, 1.0f, 0, 0, 0);
        gemm_bf16_kernel<<<dim3(S/BN, BT/BM, 1), blk, 0, stream>>>(
            v, Wki, k, nullptr, BT, S, E, 0, EPI_NONE, 1.0f, 0, 0, 0);

        rope_kernel<<<BT, 64, 0, stream>>>(q, T, S);
        rope_kernel<<<BT, 64, 0, stream>>>(k, T, S);

        // attn = sqrelu(causal(q @ k^T * scale)) : batched (T, T, S), B transposed
        gemm_bf16_kernel<<<dim3(T/BN, T/BM, Bb), blk, 0, stream>>>(
            q, k, attn, nullptr, T, T, S, 1, EPI_SQRELU_CAUSAL, attn_scale,
            (long long)T * S, (long long)T * S, (long long)T * T);

        // av = attn @ v : batched (T, E, T)
        gemm_bf16_kernel<<<dim3(E/BN, T/BM, Bb), blk, 0, stream>>>(
            attn, v, av, nullptr, T, E, T, 0, EPI_NONE, 1.0f,
            (long long)T * T, (long long)T * E, (long long)T * E);

        // g = u * av
        {
            long long n4 = (long long)BT * E / 4;
            gate_kernel<<<(unsigned)((n4 + 255) / 256), blk, 0, stream>>>(u, av, g, n4);
        }

        // x = x + g @ Wo : (BT, D, E) with residual add fused
        gemm_bf16_kernel<<<dim3(D/BN, BT/BM, 1), blk, 0, stream>>>(
            g, Woi, x, x, BT, D, E, 0, EPI_NONE, 1.0f, 0, 0, 0);
    }

    // final LN + LM head: out = ln(x) @ Wlm : (BT, V, D)
    ln_kernel<<<BT, blk, 0, stream>>>(x, lnf_g, lnf_b, xn, D);
    gemm_bf16_kernel<<<dim3(Vv/BN, BT/BM, 1), blk, 0, stream>>>(
        xn, Wlm, out, nullptr, BT, Vv, D, 0, EPI_NONE, 1.0f, 0, 0, 0);
}
